// GatingNetwork_32701880992402
// MI455X (gfx1250) — matrix operand order repaired and checked by reading, not compiled, not run on hardware
//
#include <hip/hip_runtime.h>
#include <math.h>

typedef __attribute__((ext_vector_type(16))) _Float16 v16h;
typedef __attribute__((ext_vector_type(8)))  float    v8f;

#define D_DIM 2048
#define E_DIM 64
#define T_TOK 16384
#define ROWS_PER_BLOCK 64     // 4 waves * 16 rows
#define BLOCK_THREADS 128
#define LDS_STRIDE 65         // 64 + 1 pad -> conflict-free row scans

union HVec { v16h v; _Float16 h[16]; };
union H4   { _Float16 h[4]; unsigned long long u; };

__device__ __forceinline__ void pack8(_Float16* h, float4 a, float4 b) {
  h[0] = (_Float16)a.x; h[1] = (_Float16)a.y; h[2] = (_Float16)a.z; h[3] = (_Float16)a.w;
  h[4] = (_Float16)b.x; h[5] = (_Float16)b.y; h[6] = (_Float16)b.z; h[7] = (_Float16)b.w;
}

// Convert both f32 weight matrices to f16 (row-major preserved; a lane's
// 16-half B fragment is then two aligned global_load_b128 in the main loop).
__global__ void __launch_bounds__(256)
cvt_weights(const float* __restrict__ Wg, const float* __restrict__ Wn,
            _Float16* __restrict__ Whg, _Float16* __restrict__ Whn) {
  const int i = (blockIdx.x * 256 + threadIdx.x) * 4;  // 4 elements/thread
  if (i < E_DIM * D_DIM) {
    float4 g = *(const float4*)(Wg + i);
    float4 n = *(const float4*)(Wn + i);
    H4 hg, hn;
    hg.h[0] = (_Float16)g.x; hg.h[1] = (_Float16)g.y;
    hg.h[2] = (_Float16)g.z; hg.h[3] = (_Float16)g.w;
    hn.h[0] = (_Float16)n.x; hn.h[1] = (_Float16)n.y;
    hn.h[2] = (_Float16)n.z; hn.h[3] = (_Float16)n.w;
    *(unsigned long long*)(Whg + i) = hg.u;
    *(unsigned long long*)(Whn + i) = hn.u;
  }
}

template <bool PACKED>
__global__ void __launch_bounds__(BLOCK_THREADS)
moe_gate_fused(const float* __restrict__ x,          // [T, D]
               const float* __restrict__ raw_noise,  // [T, E]
               const float* __restrict__ Wg,         // [E, D] f32
               const float* __restrict__ bg,         // [E]
               const float* __restrict__ Wn,         // [E, D] f32
               const _Float16* __restrict__ Whg,     // [E, D] f16 (PACKED)
               const _Float16* __restrict__ Whn,     // [E, D] f16 (PACKED)
               float* __restrict__ out,              // [T, E] sparse
               float* __restrict__ ws_cnt,           // [E]
               float* __restrict__ ws_psum)          // [E]
{
  __shared__ float sNoisy[ROWS_PER_BLOCK * LDS_STRIDE];
  __shared__ float sCnt[E_DIM];
  __shared__ float sPsum[E_DIM];

  const int tid  = threadIdx.x;
  const int wave = tid >> 5;
  const int lane = tid & 31;
  const int n16  = lane & 15;   // row (A) / expert-col (B,C,D) within tile
  const int kh   = lane >> 4;   // K-half selector

  if (tid < E_DIM) { sCnt[tid] = 0.f; sPsum[tid] = 0.f; }
  __syncthreads();

  const int rowBase = blockIdx.x * ROWS_PER_BLOCK + wave * 16;

  const v8f vzero = {0.f, 0.f, 0.f, 0.f, 0.f, 0.f, 0.f, 0.f};
  v8f accG[4] = {vzero, vzero, vzero, vzero};
  v8f accN[4] = {vzero, vzero, vzero, vzero};

  // -------- GEMM phase: logits & noise projections via v_wmma_f32_16x16x32_f16
  const float* xrow = x + (size_t)(rowBase + n16) * D_DIM;
  for (int kc = 0; kc < D_DIM; kc += 32) {
    // Branchless prefetch of the x stream two chunks ahead (global_prefetch_b8)
    __builtin_prefetch(xrow + ((kc + 64) & (D_DIM - 1)), 0, 3);

    // A fragment 16x32 f16: lane holds row n16; elements 0..15 <- k = kh*16+0..15
    // (contiguous order, matching the B fragment).
    HVec a;
    {
      const float4* r0 = (const float4*)(xrow + kc + kh * 16);
      const float4* r1 = (const float4*)(xrow + kc + kh * 16 + 8);
      float4 f0 = r0[0], f1 = r0[1], f2 = r1[0], f3 = r1[1];
      pack8(a.h, f0, f1);
      pack8(a.h + 8, f2, f3);
    }

    // Issue ALL 8 B-fragment loads first (distinct registers) so they fly as
    // one clause with 8 outstanding loads; WMMAs then retire against
    // staggered s_wait_loadcnt values instead of a full drain per WMMA.
    HVec bG[4], bN[4];
#pragma unroll
    for (int t = 0; t < 4; ++t) {
      const int e = t * 16 + n16;            // lane holds B column e
      const size_t boff = (size_t)e * D_DIM + kc + kh * 16; // k = kc+kh*16+i
      if constexpr (PACKED) {
        bG[t].v = *(const v16h*)(Whg + boff);  // two global_load_b128 each
        bN[t].v = *(const v16h*)(Whn + boff);
      } else {
        const float4* wg4 = (const float4*)(Wg + boff);
        const float4* wn4 = (const float4*)(Wn + boff);
        float4 g0 = wg4[0], g1 = wg4[1], g2 = wg4[2], g3 = wg4[3];
        float4 n0 = wn4[0], n1 = wn4[1], n2 = wn4[2], n3 = wn4[3];
        pack8(bG[t].h, g0, g1); pack8(bG[t].h + 8, g2, g3);
        pack8(bN[t].h, n0, n1); pack8(bN[t].h + 8, n2, n3);
      }
    }
#pragma unroll
    for (int t = 0; t < 4; ++t) {
      accG[t] = __builtin_amdgcn_wmma_f32_16x16x32_f16(
          false, a.v, false, bG[t].v, (short)0, accG[t], false, false);
      accN[t] = __builtin_amdgcn_wmma_f32_16x16x32_f16(
          false, a.v, false, bN[t].v, (short)0, accN[t], false, false);
    }
  }

  // -------- Epilogue: bias, softplus noise, noisy logits -> LDS
  // C/D layout: lanes 0-15 -> M = r, lanes 16-31 -> M = r + 8; N = n16 + t*16.
#pragma unroll
  for (int t = 0; t < 4; ++t) {
    const int e = t * 16 + n16;
    const float bias = bg[e];
#pragma unroll
    for (int r = 0; r < 8; ++r) {
      const int m = kh * 8 + r;
      const int token = rowBase + m;
      float g  = accG[t][r] + bias;
      float sn = accN[t][r];
      float sp = fmaxf(sn, 0.f) + log1pf(expf(-fabsf(sn)));  // stable softplus
      float noisy = g + raw_noise[(size_t)token * E_DIM + e] * (sp + 0.01f);
      sNoisy[(wave * 16 + m) * LDS_STRIDE + e] = noisy;
    }
  }
  __syncthreads();

  // -------- Phase 2: per-token top-2, softmaxes, sparse write, aux stats
  if (tid < ROWS_PER_BLOCK) {
    const float* rowv = &sNoisy[tid * LDS_STRIDE];
    const int token = blockIdx.x * ROWS_PER_BLOCK + tid;

    float v1 = -INFINITY, v2 = -INFINITY;
    int   i1 = 0,          i2 = 0;
    for (int e = 0; e < E_DIM; ++e) {
      float v = rowv[e];
      if (v > v1)      { v2 = v1; i2 = i1; v1 = v; i1 = e; }
      else if (v > v2) { v2 = v;  i2 = e; }
    }
    float denom = 0.f;
    for (int e = 0; e < E_DIM; ++e) denom += expf(rowv[e] - v1);
    float inv = 1.f / denom;
    for (int e = 0; e < E_DIM; ++e) atomicAdd(&sPsum[e], expf(rowv[e] - v1) * inv);
    atomicAdd(&sCnt[i1], 1.f);
    atomicAdd(&sCnt[i2], 1.f);

    float p1 = 1.f / (1.f + expf(v2 - v1));   // softmax over [top1, top2]
    float p2 = 1.f - p1;

    float* orow = out + (size_t)token * E_DIM;
    for (int e = 0; e < E_DIM; e += 4) {
      float4 z = {0.f, 0.f, 0.f, 0.f};
      if (i1 >= e && i1 < e + 4) ((float*)&z)[i1 - e] = p1;
      if (i2 >= e && i2 < e + 4) ((float*)&z)[i2 - e] = p2;
      *(float4*)(orow + e) = z;
    }
  }
  __syncthreads();

  if (tid < E_DIM) {
    atomicAdd(&ws_cnt[tid], sCnt[tid]);
    atomicAdd(&ws_psum[tid], sPsum[tid]);
  }
}

__global__ void zero_ws(float* ws) {
  if (threadIdx.x < 2 * E_DIM) ws[threadIdx.x] = 0.f;
}

__global__ void finalize_aux(const float* __restrict__ cnt,
                             const float* __restrict__ psum,
                             float* __restrict__ out) {
  if (threadIdx.x == 0) {
    float s = 0.f;
    for (int e = 0; e < E_DIM; ++e) s += cnt[e] * psum[e];
    // aux = E * sum_e (cnt_e / T) * (psum_e / T)
    out[(size_t)T_TOK * E_DIM] = s * (float)E_DIM / ((float)T_TOK * (float)T_TOK);
  }
}

extern "C" void kernel_launch(void* const* d_in, const int* in_sizes, int n_in,
                              void* d_out, int out_size, void* d_ws, size_t ws_size,
                              hipStream_t stream) {
  (void)in_sizes; (void)n_in; (void)out_size;
  const float* x  = (const float*)d_in[0];
  const float* rn = (const float*)d_in[1];
  const float* Wg = (const float*)d_in[2];
  const float* bg = (const float*)d_in[3];
  const float* Wn = (const float*)d_in[4];
  float* out = (float*)d_out;
  float* ws  = (float*)d_ws;                 // [0..63] cnt, [64..127] psum
  _Float16* Whg = (_Float16*)(ws + 2 * E_DIM);
  _Float16* Whn = Whg + (size_t)E_DIM * D_DIM;

  const size_t needed =
      2 * E_DIM * sizeof(float) + 2 * (size_t)E_DIM * D_DIM * sizeof(_Float16);

  zero_ws<<<1, 128, 0, stream>>>(ws);
  if (ws_size >= needed) {
    cvt_weights<<<(E_DIM * D_DIM / 4 + 255) / 256, 256, 0, stream>>>(Wg, Wn, Whg, Whn);
    moe_gate_fused<true><<<T_TOK / ROWS_PER_BLOCK, BLOCK_THREADS, 0, stream>>>(
        x, rn, Wg, bg, Wn, Whg, Whn, out, ws, ws + E_DIM);
  } else {
    moe_gate_fused<false><<<T_TOK / ROWS_PER_BLOCK, BLOCK_THREADS, 0, stream>>>(
        x, rn, Wg, bg, Wn, Whg, Whn, out, ws, ws + E_DIM);
  }
  finalize_aux<<<1, 32, 0, stream>>>(ws, ws + E_DIM, out);
}
